// Attention_76854144794982
// MI455X (gfx1250) — matrix operand order repaired and checked by reading, not compiled, not run on hardware
//
#include <hip/hip_runtime.h>
#include <hip/hip_bf16.h>
#include <float.h>

// CDNA5 WMMA vector types
typedef __bf16 v16bf __attribute__((ext_vector_type(16)));
typedef __bf16 v8bfv __attribute__((ext_vector_type(8)));
typedef __bf16 v4bfv __attribute__((ext_vector_type(4)));
typedef float  v8f   __attribute__((ext_vector_type(8)));

#define HH    16
#define SLEN  2048
#define DHEAD 64
#define BM    64      // query rows per block
#define BN    64      // key cols per tile
#define WAVES 4
#define NBH   32      // B*H
#define BSTRIDE 68    // padded bias row (f32) to avoid bank conflicts
#define KV_ELEMS (NBH * SLEN * DHEAD)          // 4,194,304
#define KV_BYTES ((size_t)KV_ELEMS * 2)        // 8 MB per bf16 tensor

#define LOG2E  1.4426950408889634f
#define QSCALE (0.125f * LOG2E)

#if __has_builtin(__builtin_amdgcn_exp2f)
#define EXP2F(x) __builtin_amdgcn_exp2f(x)
#else
#define EXP2F(x) exp2f(x)
#endif

// async global->LDS (CDNA5): tracked by ASYNCcnt, no VGPR staging
#if __has_builtin(__builtin_amdgcn_global_load_async_to_lds_b128) && \
    __has_builtin(__builtin_amdgcn_s_wait_asynccnt)
#define HAVE_ASYNC 1
#define NBUF 2
typedef int v4i_t __attribute__((vector_size(16)));
typedef __attribute__((address_space(1))) v4i_t* gv4i_p;
typedef __attribute__((address_space(3))) v4i_t* lv4i_p;
__device__ __forceinline__ void async_cp16(void* lds, const void* g) {
    __builtin_amdgcn_global_load_async_to_lds_b128(
        (gv4i_p)const_cast<void*>(g), (lv4i_p)lds, 0, 0);
}
__device__ __forceinline__ void async_cp16_nt(void* lds, const void* g) {
    // cpol TH=1 (non-temporal): single-use bias stream, don't evict K/V from L2
    __builtin_amdgcn_global_load_async_to_lds_b128(
        (gv4i_p)const_cast<void*>(g), (lv4i_p)lds, 0, 1);
}
#else
#define HAVE_ASYNC 0
#define NBUF 1
#endif

// xor-butterfly max over a 16-lane DPP row (VALU, no LDS traffic).
#if __has_builtin(__builtin_amdgcn_update_dpp)
#define ROWMAX16(x)                                                                        \
    do {                                                                                   \
        x = fmaxf(x, __int_as_float(__builtin_amdgcn_update_dpp(                           \
                         __float_as_int(x), __float_as_int(x), 0x161, 0xf, 0xf, true)));   \
        x = fmaxf(x, __int_as_float(__builtin_amdgcn_update_dpp(                           \
                         __float_as_int(x), __float_as_int(x), 0x162, 0xf, 0xf, true)));   \
        x = fmaxf(x, __int_as_float(__builtin_amdgcn_update_dpp(                           \
                         __float_as_int(x), __float_as_int(x), 0x164, 0xf, 0xf, true)));   \
        x = fmaxf(x, __int_as_float(__builtin_amdgcn_update_dpp(                           \
                         __float_as_int(x), __float_as_int(x), 0x168, 0xf, 0xf, true)));   \
    } while (0)
#else
#define ROWMAX16(x)                                                                        \
    do {                                                                                   \
        x = fmaxf(x, __shfl_xor(x, 1));                                                    \
        x = fmaxf(x, __shfl_xor(x, 2));                                                    \
        x = fmaxf(x, __shfl_xor(x, 4));                                                    \
        x = fmaxf(x, __shfl_xor(x, 8));                                                    \
    } while (0)
#endif

// ---------------- pre-pass 1: K (f32 row-major) -> bf16 row-major ----------------
__global__ __launch_bounds__(256)
void cvt_bf16_kernel(const float* __restrict__ in, __bf16* __restrict__ out)
{
    const int i = blockIdx.x * 256 + threadIdx.x;   // one thread = 8 elements
    const float4* p = (const float4*)in + (size_t)i * 2;
    const float4 a = p[0];
    const float4 c = p[1];
    v8bfv o;
    o[0] = (__bf16)a.x; o[1] = (__bf16)a.y; o[2] = (__bf16)a.z; o[3] = (__bf16)a.w;
    o[4] = (__bf16)c.x; o[5] = (__bf16)c.y; o[6] = (__bf16)c.z; o[7] = (__bf16)c.w;
    *(v8bfv*)(out + (size_t)i * 8) = o;
}

// -------- pre-pass 2: V (f32 [bh][s][d]) -> bf16 transposed [bh][d][s], sigma-ordered --------
// Within each 64-key tile, key j is stored at position sigma(j) = (j&15)*4 + (j>>4).
// The same permutation is applied to the staged P matrix, so P @ V is unchanged.
__global__ __launch_bounds__(256)
void transpose_v_kernel(const float* __restrict__ v, __bf16* __restrict__ vt)
{
    __shared__ __bf16 tile[32][33];
    const int s0 = blockIdx.x * 32;
    const int d0 = blockIdx.y * 32;
    const int bh = blockIdx.z;
    const int tx = threadIdx.x;          // 0..31
    const int ty0 = threadIdx.y;         // 0..7
#pragma unroll
    for (int i = 0; i < 4; ++i) {
        const int ty = ty0 + i * 8;
        tile[ty][tx] = (__bf16)v[((size_t)bh * SLEN + s0 + ty) * DHEAD + d0 + tx];
    }
    __syncthreads();
#pragma unroll
    for (int i = 0; i < 4; ++i) {
        const int ty = ty0 + i * 8;
        const int s  = s0 + tx;
        const int jl = s & 63;
        const int sp = (s & ~63) | (((jl & 15) << 2) | (jl >> 4));   // sigma within tile
        vt[((size_t)bh * DHEAD + d0 + ty) * SLEN + sp] = tile[tx][ty];
    }
}

// ---------------- main flash-attention kernel ----------------
__global__ __launch_bounds__(128)
void fattn_cdna5_kernel(const float* __restrict__ qg,
                        const __bf16* __restrict__ kbf,
                        const __bf16* __restrict__ vtbf,
                        const float* __restrict__ biasg,
                        const unsigned char* __restrict__ maskg,
                        float* __restrict__ outg)
{
    __shared__ __align__(16) __bf16 lds_k [NBUF][BN][DHEAD];     // [key][d]
    __shared__ __align__(16) __bf16 lds_vt[NBUF][DHEAD][BN];     // [d][sigma(key)]
    __shared__ __align__(16) __bf16 lds_p [WAVES][16][BN];       // P staging, sigma order
#if HAVE_ASYNC
    __shared__ __align__(16) float  lds_bias[NBUF][BM][BSTRIDE]; // bias tile (padded rows)
#endif

    const int t    = threadIdx.x;
    const int lane = t & 31;
    const int wave = t >> 5;
    const int hf   = (lane >> 4) & 1;
    const int ln   = lane & 15;

    const int bh  = blockIdx.y;          // b*H + h
    const int b   = bh >> 4;             // H == 16
    const int q0b = blockIdx.x * BM;
    const int q0  = q0b + wave * 16;

    const size_t qo_base   = (size_t)bh * SLEN * DHEAD;
    const size_t bias_base = (size_t)bh * SLEN * SLEN;

    // ---- Q once as two A-fragments (16x32 bf16), pre-scaled by 0.125*log2e ----
    const float* qrow = qg + qo_base + (size_t)(q0 + ln) * DHEAD;
    v16bf qa0, qa1;
#pragma unroll
    for (int i = 0; i < 16; ++i) {
        const int kl = i + 16 * hf;
        qa0[i] = (__bf16)(qrow[kl] * QSCALE);
        qa1[i] = (__bf16)(qrow[kl + 32] * QSCALE);
    }

    // all-ones B fragment: P @ ones -> row sums replicated to every lane
    v16bf vones;
#pragma unroll
    for (int i = 0; i < 16; ++i) vones[i] = (__bf16)1.0f;

    v8f o[4] = {};
    float m_i[8], l_i[8];
#pragma unroll
    for (int r = 0; r < 8; ++r) { m_i[r] = -FLT_MAX; l_i[r] = 0.0f; }

    // hoisted bases (loop indices become immediate offsets)
    const uint4*  gk_base   = (const uint4*)(kbf + (size_t)bh * SLEN * DHEAD);
    const __bf16* gvt_base  = vtbf + (size_t)bh * DHEAD * SLEN
                             + (size_t)(t >> 3) * SLEN + (t & 7) * 8;
    const float*  gbias_base = biasg + bias_base + (size_t)(q0b + (t >> 4)) * SLEN + (t & 15) * 4;
    const unsigned char* mrow = maskg + (size_t)b * SLEN + ln;
#if !HAVE_ASYNC
    const float*  bias_row0 = biasg + bias_base + (size_t)(q0 + 8 * hf) * SLEN + ln;
#endif

    const int niter = q0b / BN + 1;   // causal: tiles 0 .. q0b/64

#if HAVE_ASYNC
    // issue one tile (K + Vt + bias): 16 async b128 per thread
    auto issue_tile = [&](int buf, int jj) {
        const uint4* gk = gk_base + (size_t)jj * DHEAD / 8;
#pragma unroll
        for (int i = 0; i < 4; ++i)
            async_cp16(&((uint4*)&lds_k[buf][0][0])[t + 128 * i], &gk[t + 128 * i]);
        const __bf16* gvt = gvt_base + jj;
#pragma unroll
        for (int i = 0; i < 4; ++i)
            async_cp16(&lds_vt[buf][(t >> 3) + 16 * i][(t & 7) * 8], gvt + (size_t)i * 16 * SLEN);
        const float* gb = gbias_base + jj;
#pragma unroll
        for (int i = 0; i < 8; ++i)
            async_cp16_nt(&lds_bias[buf][(t >> 4) + 8 * i][(t & 15) * 4], gb + (size_t)i * 8 * SLEN);
    };
    issue_tile(0, 0);
#endif

    for (int it = 0; it < niter; ++it) {
        const int j0 = it * BN;
        const int cb = it & (NBUF - 1);
        __syncthreads();   // previous iteration's readers of the other buffer are done

#if HAVE_ASYNC
        if (it + 1 < niter) {
            issue_tile(cb ^ 1, j0 + BN);               // prefetch next tile
            __builtin_amdgcn_s_wait_asynccnt(16);      // in-order: current tile's 16 complete
        } else {
            __builtin_amdgcn_s_wait_asynccnt(0);
        }
#else
        {
            const uint4* gk = gk_base + (size_t)j0 * DHEAD / 8;
            uint4* lk = (uint4*)&lds_k[cb][0][0];
#pragma unroll
            for (int i = 0; i < 4; ++i) lk[t + 128 * i] = gk[t + 128 * i];
            const __bf16* gvt = gvt_base + j0;
#pragma unroll
            for (int i = 0; i < 4; ++i)
                *(uint4*)&lds_vt[cb][(t >> 3) + 16 * i][(t & 7) * 8] =
                    *(const uint4*)(gvt + (size_t)i * 16 * SLEN);
        }
#endif
        __syncthreads();   // everyone's current tile is resident in LDS

        // ---- matmul 1: S' = (Q*0.125*log2e) @ K^T ----
        v8f s[4] = {};
#pragma unroll
        for (int ks = 0; ks < 2; ++ks) {
            const v16bf aa = ks ? qa1 : qa0;
#pragma unroll
            for (int tN = 0; tN < 4; ++tN) {
                v16bf bk;
#pragma unroll
                for (int i = 0; i < 16; ++i)
                    bk[i] = lds_k[cb][tN * 16 + ln][i + 16 * hf + 32 * ks];
                s[tN] = __builtin_amdgcn_wmma_f32_16x16x32_bf16(false, aa, false, bk, (short)0, s[tN], false, false);
            }
        }

        // ---- add bias*log2e, stay in log2 domain ----
#if HAVE_ASYNC
        {
            const float* bb = &lds_bias[cb][wave * 16 + 8 * hf][ln];
#pragma unroll
            for (int r = 0; r < 8; ++r)
#pragma unroll
                for (int tN = 0; tN < 4; ++tN)
                    s[tN][r] = fmaf(bb[r * BSTRIDE + tN * 16], LOG2E, s[tN][r]);
        }
#else
        {
            const float* bp = bias_row0 + j0;
#pragma unroll
            for (int r = 0; r < 8; ++r)
#pragma unroll
                for (int tN = 0; tN < 4; ++tN)
                    s[tN][r] = fmaf(__builtin_nontemporal_load(bp + (size_t)r * SLEN + 16 * tN),
                                    LOG2E, s[tN][r]);
        }
#endif

        // ---- key-padding mask: skip entirely when all keys kept (uniform branch) ----
        bool km[4];
#pragma unroll
        for (int tN = 0; tN < 4; ++tN) km[tN] = mrow[j0 + 16 * tN] != 0;
        if (!__all(km[0] && km[1] && km[2] && km[3])) {
#pragma unroll
            for (int tN = 0; tN < 4; ++tN)
                if (!km[tN])
#pragma unroll
                    for (int r = 0; r < 8; ++r) s[tN][r] = -FLT_MAX;
        }

        // ---- causal mask: only the diagonal tile needs it (uniform branch) ----
        if (it == niter - 1) {
#pragma unroll
            for (int r = 0; r < 8; ++r) {
                const int row = q0 + r + 8 * hf;
#pragma unroll
                for (int tN = 0; tN < 4; ++tN)
                    if (j0 + tN * 16 + ln > row) s[tN][r] = -FLT_MAX;
            }
        }

        // ---- online max (DPP xor-butterfly) + exp2 + stage P (sigma order, b64 stores) ----
        float alpha[8];
#pragma unroll
        for (int r = 0; r < 8; ++r) {
            float mx = fmaxf(fmaxf(s[0][r], s[1][r]), fmaxf(s[2][r], s[3][r]));
            ROWMAX16(mx);
            const float nm = fmaxf(m_i[r], mx);
            alpha[r] = EXP2F(m_i[r] - nm);
            m_i[r]   = nm;
            v4bfv pq;
            pq[0] = (__bf16)EXP2F(s[0][r] - nm);
            pq[1] = (__bf16)EXP2F(s[1][r] - nm);
            pq[2] = (__bf16)EXP2F(s[2][r] - nm);
            pq[3] = (__bf16)EXP2F(s[3][r] - nm);
            // key j = tN*16+ln lands at sigma(j) = ln*4 + tN  -> contiguous 4 bf16
            *(v4bfv*)&lds_p[wave][r + 8 * hf][ln * 4] = pq;
        }

        // ---- rescale running O and l ----
#pragma unroll
        for (int r = 0; r < 8; ++r) {
            o[0][r] *= alpha[r]; o[1][r] *= alpha[r];
            o[2][r] *= alpha[r]; o[3][r] *= alpha[r];
            l_i[r]  *= alpha[r];
        }

        // ---- P back in A-layout over sigma positions (DS in-order within wave) ----
        v16bf pa0, pa1;
#pragma unroll
        for (int i = 0; i < 16; ++i) {
            const int kl = i + 16 * hf;
            pa0[i] = lds_p[wave][ln][kl];
            pa1[i] = lds_p[wave][ln][kl + 32];
        }

        // ---- row sums of P via ones-matmul (permutation-invariant) ----
        v8f rs = {};
        rs = __builtin_amdgcn_wmma_f32_16x16x32_bf16(false, pa0, false, vones, (short)0, rs, false, false);
        rs = __builtin_amdgcn_wmma_f32_16x16x32_bf16(false, pa1, false, vones, (short)0, rs, false, false);
#pragma unroll
        for (int r = 0; r < 8; ++r) l_i[r] += rs[r];

        // ---- matmul 2: O += P @ V (sigma-consistent K order) ----
#pragma unroll
        for (int ks = 0; ks < 2; ++ks) {
            const v16bf pa = ks ? pa1 : pa0;
#pragma unroll
            for (int dt = 0; dt < 4; ++dt) {
                v16bf bv;
#pragma unroll
                for (int i = 0; i < 16; ++i)
                    bv[i] = lds_vt[cb][dt * 16 + ln][i + 16 * hf + 32 * ks];
                o[dt] = __builtin_amdgcn_wmma_f32_16x16x32_bf16(false, pa, false, bv, (short)0, o[dt], false, false);
            }
        }
    }

    // ---- finalize: O / l, non-temporal stores ----
    float rl[8];
#pragma unroll
    for (int r = 0; r < 8; ++r) rl[r] = 1.0f / l_i[r];
#pragma unroll
    for (int dt = 0; dt < 4; ++dt) {
#pragma unroll
        for (int r = 0; r < 8; ++r) {
            float* op = outg + qo_base + (size_t)(q0 + r + 8 * hf) * DHEAD + dt * 16 + ln;
            __builtin_nontemporal_store(o[dt][r] * rl[r], op);
        }
    }
}

extern "C" void kernel_launch(void* const* d_in, const int* in_sizes, int n_in,
                              void* d_out, int out_size, void* d_ws, size_t ws_size,
                              hipStream_t stream) {
    const float*         q    = (const float*)d_in[0];
    const float*         k    = (const float*)d_in[1];
    const float*         v    = (const float*)d_in[2];
    const float*         bias = (const float*)d_in[3];
    const unsigned char* mask = (const unsigned char*)d_in[4];
    float*               out  = (float*)d_out;

    __bf16* kbf = (__bf16*)d_ws;
    __bf16* vtb = (__bf16*)((char*)d_ws + KV_BYTES);

    // pre-pass: K -> bf16, V -> bf16 transposed [bh][d][s] (sigma-ordered per 64-key tile)
    hipLaunchKernelGGL(cvt_bf16_kernel, dim3(KV_ELEMS / (256 * 8)), dim3(256), 0, stream, k, kbf);
    hipLaunchKernelGGL(transpose_v_kernel, dim3(SLEN / 32, DHEAD / 32, NBH), dim3(32, 8), 0, stream, v, vtb);

    // main kernel
    hipLaunchKernelGGL(fattn_cdna5_kernel, dim3(SLEN / BM, NBH), dim3(128), 0, stream,
                       q, kbf, vtb, bias, mask, out);
}
